// PatchCaps_70875550318605
// MI455X (gfx1250) — matrix operand order repaired and checked by reading, not compiled, not run on hardware
//
#include <hip/hip_runtime.h>

// ---------------------------------------------------------------------------
// Types
// ---------------------------------------------------------------------------
typedef __bf16 bf16_t;
typedef __attribute__((ext_vector_type(16))) __bf16 v16bf;
typedef __attribute__((ext_vector_type(8)))  __bf16 v8bf;
typedef __attribute__((ext_vector_type(8)))  float  v8f;
typedef __attribute__((ext_vector_type(4)))  int    v4i;

// AS-qualified pointee typedefs for the async-copy builtin's prototype
typedef __attribute__((address_space(1))) v4i as1_v4i;   // global
typedef __attribute__((address_space(3))) v4i as3_v4i;   // LDS

// ---------------------------------------------------------------------------
// CDNA5 async global->LDS copy (guarded so it compiles on any toolchain).
// ---------------------------------------------------------------------------
#if defined(__gfx1250__) && __has_builtin(__builtin_amdgcn_global_load_async_to_lds_b128)
#define HAS_ASYNC_LDS 1
#endif

__device__ __forceinline__ void cp_async16(bf16_t* lds_dst, const bf16_t* gsrc) {
#ifdef HAS_ASYNC_LDS
  __builtin_amdgcn_global_load_async_to_lds_b128(
      (as1_v4i*)gsrc, (as3_v4i*)lds_dst, 0, 0);
#else
  *(v8bf*)lds_dst = *(const v8bf*)gsrc;
#endif
}

__device__ __forceinline__ void async_wait0() {
#ifdef HAS_ASYNC_LDS
#if __has_builtin(__builtin_amdgcn_s_wait_asynccnt)
  __builtin_amdgcn_s_wait_asynccnt(0);
#else
  asm volatile("s_wait_asynccnt 0" ::: "memory");
#endif
#endif
}
__device__ __forceinline__ void async_wait4() {
#ifdef HAS_ASYNC_LDS
#if __has_builtin(__builtin_amdgcn_s_wait_asynccnt)
  __builtin_amdgcn_s_wait_asynccnt(4);
#else
  asm volatile("s_wait_asynccnt 4" ::: "memory");
#endif
#endif
}

// ---------------------------------------------------------------------------
// Core: bf16 WMMA GEMM.  C[M,N] = act( A[M,Kp] * Bt[N,Kp]^T * scale + shift + res )
//   A  : bf16 row-major, lda = Kp   (M multiple of 64, Kp multiple of 32)
//   Bt : bf16 row-major [Npad, Kp]  (B transposed; weights are stored [N,K]; Npad mult 64)
//   Epilogue: per-column scale/shift (BN / bias), optional residual, act:
//             0 = none, 1 = relu, 2 = quickgelu
// Block tile 64x64, BK=32, 4 waves; each wave owns a 32x32 quadrant computed as
// a 2x2 grid of 16x16 WMMA accumulators (every fragment feeds 2 WMMAs).
// Double-buffered LDS fed by async global->LDS copies (4 x b128 per thread/stage).
// ---------------------------------------------------------------------------
__global__ __launch_bounds__(128)
void gemm_bf16_wmma_kernel(const bf16_t* __restrict__ A, int lda,
                           const bf16_t* __restrict__ Bt, int ldb,
                           float* __restrict__ C, int ldc,
                           const float* __restrict__ scale,
                           const float* __restrict__ shift,
                           const float* __restrict__ res, int ldr,
                           int M, int N, int Kp, int actmode) {
  __shared__ __align__(16) bf16_t As[2][64][32];
  __shared__ __align__(16) bf16_t Bs[2][64][32];

  const int t   = threadIdx.x;
  const int row = t >> 2;          // 0..31 : first tile row loaded by this thread
  const int ch  = t & 3;           // 0..3  : 16-byte chunk within the 64B row

  const long long rowA = (long long)blockIdx.y * 64 + row;
  const long long rowB = (long long)blockIdx.x * 64 + row;
  const bf16_t* ga0 = A  + rowA * lda + ch * 8;
  const bf16_t* ga1 = ga0 + 32LL * lda;          // rows +32
  const bf16_t* gb0 = Bt + rowB * ldb + ch * 8;
  const bf16_t* gb1 = gb0 + 32LL * ldb;

  const int wv   = t >> 5;                 // wave id 0..3
  const int lane = t & 31;
  const int wy   = (wv >> 1) * 32;         // wave quadrant row offset
  const int wx   = (wv & 1) * 32;          // wave quadrant col offset
  const int mrow = lane & 15;
  const int half = lane >> 4;              // 0: lanes 0-15, 1: lanes 16-31

  v8f acc[2][2];
#pragma unroll
  for (int ri = 0; ri < 2; ++ri)
#pragma unroll
    for (int ci = 0; ci < 2; ++ci)
#pragma unroll
      for (int i = 0; i < 8; ++i) acc[ri][ci][i] = 0.0f;

  const int nk = Kp >> 5;

  auto stage = [&](int buf, int kk) {
    cp_async16(&As[buf][row][ch * 8],      ga0 + kk);
    cp_async16(&As[buf][row + 32][ch * 8], ga1 + kk);
    cp_async16(&Bs[buf][row][ch * 8],      gb0 + kk);
    cp_async16(&Bs[buf][row + 32][ch * 8], gb1 + kk);
  };

  auto compute = [&](int cur) {
    // A fragments (16x32 bf16): lane<16 holds K 0..7 & 16..23, lane>=16 holds 8..15 & 24..31
    v16bf af[2];
#pragma unroll
    for (int ri = 0; ri < 2; ++ri) {
      const int ar = wy + ri * 16 + mrow;
      v8bf lo = *(const v8bf*)&As[cur][ar][half * 8];
      v8bf hi = *(const v8bf*)&As[cur][ar][half * 8 + 16];
#pragma unroll
      for (int i = 0; i < 8; ++i) { af[ri][i] = lo[i]; af[ri][i + 8] = hi[i]; }
    }
    // B fragments (32x16 bf16, N-major rows): lane<16 holds K 0..7 & 16..23, lane>=16 holds 8..15 & 24..31
    v16bf bfr[2];
#pragma unroll
    for (int ci = 0; ci < 2; ++ci) {
      const int br = wx + ci * 16 + mrow;
      v8bf lo = *(const v8bf*)&Bs[cur][br][half * 8];
      v8bf hi = *(const v8bf*)&Bs[cur][br][half * 8 + 16];
#pragma unroll
      for (int i = 0; i < 8; ++i) { bfr[ci][i] = lo[i]; bfr[ci][i + 8] = hi[i]; }
    }
#pragma unroll
    for (int ri = 0; ri < 2; ++ri)
#pragma unroll
      for (int ci = 0; ci < 2; ++ci)
        acc[ri][ci] = __builtin_amdgcn_wmma_f32_16x16x32_bf16(
            false, af[ri], false, bfr[ci], (short)0, acc[ri][ci], false, false);
  };

  // stage tile 0
  stage(0, 0);

  // steady state: branch-free body (last iteration peeled)
  for (int tk = 0; tk < nk - 1; ++tk) {
    const int cur = tk & 1;
    const int kk = (tk + 1) * 32;
    __builtin_prefetch(ga0 + kk + 32, 0, 1);     // global_prefetch for tk+2
    __builtin_prefetch(gb0 + kk + 32, 0, 1);
    stage(cur ^ 1, kk);
    async_wait4();                               // current tile landed
    __syncthreads();
    compute(cur);
    __syncthreads();                             // done reading 'cur' before restage
  }
  async_wait0();
  __syncthreads();
  compute((nk - 1) & 1);

  // Epilogue: C/D layout: VGPR r -> row (r + 8*half), col = lane&15
#pragma unroll
  for (int ci = 0; ci < 2; ++ci) {
    const long long gcol = (long long)blockIdx.x * 64 + wx + ci * 16 + (lane & 15);
    if (gcol < N) {
      const float sc = scale ? scale[gcol] : 1.0f;
      const float sh = shift ? shift[gcol] : 0.0f;
#pragma unroll
      for (int ri = 0; ri < 2; ++ri) {
#pragma unroll
        for (int r = 0; r < 8; ++r) {
          const long long grow = (long long)blockIdx.y * 64 + wy + ri * 16 + r + 8 * half;
          if (grow < M) {
            float v = acc[ri][ci][r] * sc + sh;
            if (res) v += res[grow * (long long)ldr + gcol];
            if (actmode == 1)      v = fmaxf(v, 0.0f);
            else if (actmode == 2) v = v / (1.0f + __expf(-1.702f * v));
            C[grow * (long long)ldc + gcol] = v;
          }
        }
      }
    }
  }
}

// ---------------------------------------------------------------------------
// im2col: gather f32 activations (NCHW or NHWC) into bf16 [Mp, Kp], zero pad.
// K index order = ci*k*k + kh*k + kw (matches OIHW weight flatten).
// ---------------------------------------------------------------------------
__global__ void im2col_kernel(const float* __restrict__ src, bf16_t* __restrict__ dst,
                              int nchw, int Bn, int Cc, int H, int W,
                              int k, int stride, int pad, int OH, int OW,
                              long long Mp, int Kp) {
  long long idx = (long long)blockIdx.x * blockDim.x + threadIdx.x;
  long long total = Mp * (long long)Kp;
  if (idx >= total) return;
  long long m = idx / Kp;
  int j = (int)(idx - m * Kp);
  float v = 0.0f;
  long long M = (long long)Bn * OH * OW;
  int KK = Cc * k * k;
  if (m < M && j < KK) {
    int b  = (int)(m / (OH * OW));
    int r  = (int)(m % (OH * OW));
    int oh = r / OW, ow = r % OW;
    int ci = j / (k * k);
    int rr = j % (k * k);
    int kh = rr / k, kw = rr % k;
    int ih = oh * stride - pad + kh;
    int iw = ow * stride - pad + kw;
    if (ih >= 0 && ih < H && iw >= 0 && iw < W) {
      v = nchw ? src[(((long long)b * Cc + ci) * H + ih) * W + iw]
               : src[(((long long)b * H + ih) * W + iw) * Cc + ci];
    }
  }
  dst[idx] = (bf16_t)v;
}

// ---------------------------------------------------------------------------
// f32 [rows?, lds] -> bf16 [Np, Kp], copy cols<K & rows<Nvalid, zero-pad rest.
// Used both for weight downconvert ([N,K] -> padded) and activation repack.
// ---------------------------------------------------------------------------
__global__ void cast_pad_kernel(const float* __restrict__ src, bf16_t* __restrict__ dst,
                                long long Nvalid, int lds, int K, int Kp, long long Np) {
  long long idx = (long long)blockIdx.x * blockDim.x + threadIdx.x;
  long long total = Np * (long long)Kp;
  if (idx >= total) return;
  long long m = idx / Kp;
  int kk = (int)(idx - m * Kp);
  float v = (m < Nvalid && kk < K) ? src[m * (long long)lds + kk] : 0.0f;
  dst[idx] = (bf16_t)v;
}

// ---------------------------------------------------------------------------
// Fold conv-bias + batchnorm into per-channel scale/shift.
// y = (conv + b - mean) * gamma*rsqrt(var+eps) + beta  = conv*s + shift
// ---------------------------------------------------------------------------
__global__ void bn_coeff_kernel(const float* bias, const float* gamma, const float* beta,
                                const float* mean, const float* var,
                                float* scale, float* shift, int C) {
  int n = blockIdx.x * blockDim.x + threadIdx.x;
  if (n < C) {
    float s = gamma[n] * rsqrtf(var[n] + 1e-5f);
    scale[n] = s;
    shift[n] = beta[n] + (bias[n] - mean[n]) * s;
  }
}

// ---------------------------------------------------------------------------
// h[b,p,:] += pos[p,:]
// ---------------------------------------------------------------------------
__global__ void add_pos_kernel(float* h, const float* pos, int Bn, int P, int D) {
  long long idx = (long long)blockIdx.x * blockDim.x + threadIdx.x;
  long long total = (long long)Bn * P * D;
  if (idx >= total) return;
  int d = (int)(idx % D);
  int p = (int)((idx / D) % P);
  h[idx] += pos[p * D + d];
}

// ---------------------------------------------------------------------------
// LayerNorm over last dim D (=48), one row per block (64 threads).
// Optionally writes f32 out and/or zero-padded bf16 out (for GEMM A operand).
// ---------------------------------------------------------------------------
__global__ __launch_bounds__(64)
void ln_kernel(const float* __restrict__ x, const float* __restrict__ g,
               const float* __restrict__ b, float* __restrict__ outf,
               bf16_t* __restrict__ outb, int D, int Kp) {
  const long long row = blockIdx.x;
  const int t = threadIdx.x;
  __shared__ float red[64];
  float v = (t < D) ? x[row * D + t] : 0.0f;
  red[t] = v;
  __syncthreads();
  for (int s = 32; s > 0; s >>= 1) { if (t < s) red[t] += red[t + s]; __syncthreads(); }
  float mean = red[0] / (float)D;
  __syncthreads();
  float d = (t < D) ? (v - mean) : 0.0f;
  red[t] = d * d;
  __syncthreads();
  for (int s = 32; s > 0; s >>= 1) { if (t < s) red[t] += red[t + s]; __syncthreads(); }
  float inv = rsqrtf(red[0] / (float)D + 1e-5f);
  float y = (t < D) ? (d * inv * g[t] + b[t]) : 0.0f;
  if (outf && t < D) outf[row * D + t] = y;
  if (outb && t < Kp) outb[row * (long long)Kp + t] = (bf16_t)y;
}

// ---------------------------------------------------------------------------
// MHA core for head_dim = 4, D = 48, S <= 64.  One block per (batch, head).
// qkv layout: [B*S, ldq] with q at col h*4, k at 48+h*4, v at 96+h*4.
// ---------------------------------------------------------------------------
__global__ __launch_bounds__(64)
void attn_kernel(const float* __restrict__ qkv, int ldq,
                 float* __restrict__ out, int ldo, int S, int H) {
  const int D = 48;
  const int b = blockIdx.x / H;
  const int h = blockIdx.x % H;
  const int s = threadIdx.x;
  __shared__ float Ks[64][4];
  __shared__ float Vs[64][4];
  if (s < S) {
    const float* base = qkv + ((long long)(b * S + s)) * ldq;
#pragma unroll
    for (int e = 0; e < 4; ++e) {
      Ks[s][e] = base[D     + h * 4 + e];
      Vs[s][e] = base[2 * D + h * 4 + e];
    }
  }
  __syncthreads();
  if (s >= S) return;
  const float* qb = qkv + ((long long)(b * S + s)) * ldq + h * 4;
  const float q0 = qb[0], q1 = qb[1], q2 = qb[2], q3 = qb[3];
  float sc[64];
  float mx = -1e30f;
  for (int j = 0; j < S; ++j) {
    float d = (q0 * Ks[j][0] + q1 * Ks[j][1] + q2 * Ks[j][2] + q3 * Ks[j][3]) * 0.5f;
    sc[j] = d;
    mx = fmaxf(mx, d);
  }
  float sum = 0.0f, a0 = 0.0f, a1 = 0.0f, a2 = 0.0f, a3 = 0.0f;
  for (int j = 0; j < S; ++j) {
    float p = __expf(sc[j] - mx);
    sum += p;
    a0 += p * Vs[j][0]; a1 += p * Vs[j][1]; a2 += p * Vs[j][2]; a3 += p * Vs[j][3];
  }
  float r = 1.0f / sum;
  float* ob = out + ((long long)(b * S + s)) * ldo + h * 4;
  ob[0] = a0 * r; ob[1] = a1 * r; ob[2] = a2 * r; ob[3] = a3 * r;
}

// ---------------------------------------------------------------------------
// Capsule routing: one block per (batch, out-capsule k).  n0 <= 64, D = 48.
//   U[i,l] = sum_j x[b,i,j] * W[k,i,j,l]
//   A[i,j] = <U[i],U[j]> / sqrt(48); C = softmax_j(A) + bco[k,i,j]
//   S[l] = sum_i U[i,l] * sum_j C[i,j];  out = squash(S)
// ---------------------------------------------------------------------------
__global__ __launch_bounds__(64)
void routing_kernel(const float* __restrict__ x, const float* __restrict__ W,
                    const float* __restrict__ bco, float* __restrict__ out,
                    int n0, int n1, int rows_per_b, int row_off) {
  const int D = 48;
  const int b = blockIdx.x / n1;
  const int k = blockIdx.x % n1;
  const int t = threadIdx.x;
  __shared__ float U[64 * 48];
  __shared__ float Cs[64];
  __shared__ float Sv[48];
  __shared__ float red[64];

  for (int e = t; e < n0 * D; e += 64) {
    int i = e / D, l = e % D;
    const float* xr = x + ((long long)(b * n0 + i)) * D;
    const float* wr = W + (((long long)(k * n0 + i)) * D) * D + l;  // stride D over j
    float s = 0.0f;
    for (int j = 0; j < D; ++j) s += xr[j] * wr[(long long)j * D];
    U[i * D + l] = s;
  }
  __syncthreads();

  const float inv = rsqrtf(48.0f);
  for (int i = t; i < n0; i += 64) {
    float a[64];
    float mx = -1e30f;
    for (int j = 0; j < n0; ++j) {
      float d = 0.0f;
      for (int l = 0; l < D; ++l) d += U[i * D + l] * U[j * D + l];
      d *= inv;
      a[j] = d;
      mx = fmaxf(mx, d);
    }
    float sum = 0.0f;
    for (int j = 0; j < n0; ++j) { a[j] = __expf(a[j] - mx); sum += a[j]; }
    float cs = 0.0f;
    const float* brow = bco + ((long long)(k * n0 + i)) * n0;
    for (int j = 0; j < n0; ++j) cs += a[j] / sum + brow[j];
    Cs[i] = cs;
  }
  __syncthreads();

  if (t < D) {
    float s = 0.0f;
    for (int i = 0; i < n0; ++i) s += U[i * D + t] * Cs[i];
    Sv[t] = s;
  }
  __syncthreads();
  red[t] = (t < D) ? Sv[t] * Sv[t] : 0.0f;
  __syncthreads();
  for (int s2 = 32; s2 > 0; s2 >>= 1) { if (t < s2) red[t] += red[t + s2]; __syncthreads(); }
  const float sq = red[0];
  if (t < D) {
    float val = sq / (1.0f + sq) * Sv[t] / (sqrtf(sq) + 1e-5f);
    out[((long long)b * rows_per_b + row_off + k) * D + t] = val;
  }
}

// ---------------------------------------------------------------------------
// Host orchestration
// ---------------------------------------------------------------------------
static inline long long rup(long long v, long long m) { return (v + m - 1) / m * m; }

extern "C" void kernel_launch(void* const* d_in, const int* in_sizes, int n_in,
                              void* d_out, int out_size, void* d_ws, size_t ws_size,
                              hipStream_t stream) {
  (void)in_sizes; (void)n_in; (void)out_size; (void)ws_size;

  // ---- input unpack (setup_inputs dict order, recursive insertion order) ----
  int gi = 0;
  auto F = [&](int i) { return (const float*)d_in[i]; };
  const float* X = F(gi++);  // [64,3,32,32]

  struct ConvP { const float *w, *b, *g, *be, *mn, *vr; };
  struct BlkP  { const float *l1s, *l1b, *inw, *inb, *ow, *ob, *l2s, *l2b, *fw, *fb, *pw, *pb; };
  struct PatP  { const float *pw, *pb, *pos, *l1s, *l1b, *l2s, *l2b; BlkP blk[2]; };
  ConvP convs[4][4];
  PatP  pat[4];
  for (int bi = 0; bi < 4; ++bi) {
    for (int ci = 0; ci < 4; ++ci) {
      convs[bi][ci] = { F(gi), F(gi + 1), F(gi + 2), F(gi + 3), F(gi + 4), F(gi + 5) };
      gi += 6;
    }
    PatP& p = pat[bi];
    p.pw = F(gi++); p.pb = F(gi++); p.pos = F(gi++);
    p.l1s = F(gi++); p.l1b = F(gi++); p.l2s = F(gi++); p.l2b = F(gi++);
    for (int li = 0; li < 2; ++li) {
      BlkP& q = p.blk[li];
      q.l1s = F(gi++); q.l1b = F(gi++); q.inw = F(gi++); q.inb = F(gi++);
      q.ow  = F(gi++); q.ob  = F(gi++); q.l2s = F(gi++); q.l2b = F(gi++);
      q.fw  = F(gi++); q.fb  = F(gi++); q.pw  = F(gi++); q.pb  = F(gi++);
    }
  }
  const float* rW[5]; const float* rb[5];
  for (int i = 0; i < 5; ++i) { rW[i] = F(gi++); rb[i] = F(gi++); }

  // ---- workspace bump allocator ----
  char* wsp = (char*)d_ws;
  size_t off = 0;
  auto alloc = [&](size_t bytes) -> void* {
    void* p = wsp + off;
    off += (bytes + 255) & ~(size_t)255;
    return p;
  };
  bf16_t* abuf = (bf16_t*)alloc((size_t)65536 * 1152 * sizeof(bf16_t)); // A operand (im2col / tokens)
  bf16_t* wbuf = (bf16_t*)alloc((size_t)256 * 4096 * sizeof(bf16_t));   // weight operand
  float*  actP = (float*)alloc((size_t)65536 * 256 * sizeof(float));
  float*  actQ = (float*)alloc((size_t)65536 * 256 * sizeof(float));
  float*  tok  = (float*)alloc((size_t)4096 * 48 * sizeof(float));
  float*  tmp1 = (float*)alloc((size_t)4096 * 48 * sizeof(float));
  float*  tmp2 = (float*)alloc((size_t)4096 * 48 * sizeof(float));
  float*  qkvb = (float*)alloc((size_t)4096 * 144 * sizeof(float));
  float*  mlpb = (float*)alloc((size_t)4096 * 192 * sizeof(float));
  float*  bnsc = (float*)alloc(256 * sizeof(float));
  float*  bnsh = (float*)alloc(256 * sizeof(float));
  float*  capc = (float*)alloc((size_t)64 * 43 * 48 * sizeof(float));

  // ---- helpers ----
  auto launch_gemm = [&](const bf16_t* A, int lda, const bf16_t* Bt, int ldb,
                         float* C, int ldc, const float* sc, const float* sh,
                         const float* res, int ldr, long long M, int N, int Kp, int act) {
    dim3 grid((unsigned)(rup(N, 64) / 64), (unsigned)(M / 64));
    gemm_bf16_wmma_kernel<<<grid, 128, 0, stream>>>(A, lda, Bt, ldb, C, ldc,
                                                    sc, sh, res, ldr, (int)M, N, Kp, act);
  };
  auto launch_wcast = [&](const float* w, int N, int K, int Kp) {
    long long Np = rup(N, 64);
    long long total = Np * (long long)Kp;
    cast_pad_kernel<<<(unsigned)((total + 255) / 256), 256, 0, stream>>>(w, wbuf, N, K, K, Kp, Np);
  };
  auto launch_acast = [&](const float* a, long long M, int lds, int K, int Kp) {
    long long total = M * (long long)Kp;
    cast_pad_kernel<<<(unsigned)((total + 255) / 256), 256, 0, stream>>>(a, abuf, M, lds, K, Kp, M);
  };
  auto launch_im2col = [&](const float* src, int nchw, int Bn, int Cc, int H, int W,
                           int k, int s, int p, int OH, int OW, long long M, int Kp) {
    long long total = M * (long long)Kp;
    im2col_kernel<<<(unsigned)((total + 255) / 256), 256, 0, stream>>>(
        src, abuf, nchw, Bn, Cc, H, W, k, s, p, OH, OW, M, Kp);
  };

  static const int STR[4]  = {1, 2, 4, 8};
  static const int N1S[4]  = {32, 8, 2, 1};
  static const int OFFS[4] = {0, 32, 40, 42};

  for (int bi = 0; bi < 4; ++bi) {
    const int s1 = STR[bi];
    const int H1 = 31 / s1 + 1;                 // conv1 output spatial (32,16,8,4)
    const long long Mc = 64LL * H1 * H1;        // rows of conv GEMMs (mult of 64)

    // conv1: 3->64, k5, stride s1, pad 2, from NCHW f32 input
    {
      const int K = 3 * 25, Kp = 96, Cout = 64;
      launch_im2col(X, 1, 64, 3, 32, 32, 5, s1, 2, H1, H1, Mc, Kp);
      launch_wcast(convs[bi][0].w, Cout, K, Kp);
      bn_coeff_kernel<<<1, 256, 0, stream>>>(convs[bi][0].b, convs[bi][0].g, convs[bi][0].be,
                                             convs[bi][0].mn, convs[bi][0].vr, bnsc, bnsh, Cout);
      launch_gemm(abuf, Kp, wbuf, Kp, actP, Cout, bnsc, bnsh, nullptr, 0, Mc, Cout, Kp, 1);
    }
    // conv2: 64->128, k3 s1 p1, NHWC activations
    {
      const int K = 64 * 9, Kp = 576, Cout = 128;
      launch_im2col(actP, 0, 64, 64, H1, H1, 3, 1, 1, H1, H1, Mc, Kp);
      launch_wcast(convs[bi][1].w, Cout, K, Kp);
      bn_coeff_kernel<<<1, 256, 0, stream>>>(convs[bi][1].b, convs[bi][1].g, convs[bi][1].be,
                                             convs[bi][1].mn, convs[bi][1].vr, bnsc, bnsh, Cout);
      launch_gemm(abuf, Kp, wbuf, Kp, actQ, Cout, bnsc, bnsh, nullptr, 0, Mc, Cout, Kp, 1);
    }
    // conv3: 128->128 with residual (pre-relu)
    {
      const int K = 128 * 9, Kp = 1152, Cout = 128;
      launch_im2col(actQ, 0, 64, 128, H1, H1, 3, 1, 1, H1, H1, Mc, Kp);
      launch_wcast(convs[bi][2].w, Cout, K, Kp);
      bn_coeff_kernel<<<1, 256, 0, stream>>>(convs[bi][2].b, convs[bi][2].g, convs[bi][2].be,
                                             convs[bi][2].mn, convs[bi][2].vr, bnsc, bnsh, Cout);
      launch_gemm(abuf, Kp, wbuf, Kp, actP, Cout, bnsc, bnsh, actQ, 128, Mc, Cout, Kp, 1);
    }
    // conv4: 128->256
    {
      const int K = 128 * 9, Kp = 1152, Cout = 256;
      launch_im2col(actP, 0, 64, 128, H1, H1, 3, 1, 1, H1, H1, Mc, Kp);
      launch_wcast(convs[bi][3].w, Cout, K, Kp);
      bn_coeff_kernel<<<1, 256, 0, stream>>>(convs[bi][3].b, convs[bi][3].g, convs[bi][3].be,
                                             convs[bi][3].mn, convs[bi][3].vr, bnsc, bnsh, Cout);
      launch_gemm(abuf, Kp, wbuf, Kp, actQ, Cout, bnsc, bnsh, nullptr, 0, Mc, Cout, Kp, 1);
    }

    // patchify: conv 4x4 stride 4 -> tokens [B*P, 48]
    const int Pside = H1 / 4;
    const int P = Pside * Pside;
    const long long Mt = 64LL * P;
    {
      const int Kp = 4096;   // 256*4*4
      launch_im2col(actQ, 0, 64, 256, H1, H1, 4, 4, 0, Pside, Pside, Mt, Kp);
      launch_wcast(pat[bi].pw, 48, Kp, Kp);
      launch_gemm(abuf, Kp, wbuf, Kp, tok, 48, nullptr, pat[bi].pb, nullptr, 0, Mt, 48, Kp, 0);
    }
    {
      long long total = Mt * 48;
      add_pos_kernel<<<(unsigned)((total + 255) / 256), 256, 0, stream>>>(tok, pat[bi].pos, 64, P, 48);
    }
    // pre-block layernorm -> h (tmp1, f32)
    ln_kernel<<<(unsigned)Mt, 64, 0, stream>>>(tok, pat[bi].l1s, pat[bi].l1b, tmp1, nullptr, 48, 64);

    // 2 encoder blocks; h lives in tmp1, x lives in tok
    for (int li = 0; li < 2; ++li) {
      const BlkP& q = pat[bi].blk[li];
      // ln1(h) -> bf16 A
      ln_kernel<<<(unsigned)Mt, 64, 0, stream>>>(tmp1, q.l1s, q.l1b, nullptr, abuf, 48, 64);
      // qkv = ln1(h) @ in_w.T + in_b   [Mt, 144]
      launch_wcast(q.inw, 144, 48, 64);
      launch_gemm(abuf, 64, wbuf, 64, qkvb, 144, nullptr, q.inb, nullptr, 0, Mt, 144, 64, 0);
      // attention -> tmp2 [Mt, 48]
      attn_kernel<<<64 * 12, 64, 0, stream>>>(qkvb, 144, tmp2, 48, P, 12);
      // out proj + residual(h) -> x (tok)
      launch_acast(tmp2, Mt, 48, 48, 64);
      launch_wcast(q.ow, 48, 48, 64);
      launch_gemm(abuf, 64, wbuf, 64, tok, 48, nullptr, q.ob, tmp1, 48, Mt, 48, 64, 0);
      // ln2(x) -> bf16 A
      ln_kernel<<<(unsigned)Mt, 64, 0, stream>>>(tok, q.l2s, q.l2b, nullptr, abuf, 48, 64);
      // fc + quickgelu -> mlpb [Mt, 192]
      launch_wcast(q.fw, 192, 48, 64);
      launch_gemm(abuf, 64, wbuf, 64, mlpb, 192, nullptr, q.fb, nullptr, 0, Mt, 192, 64, 2);
      // proj + residual(x) -> h (tmp1)
      launch_acast(mlpb, Mt, 192, 192, 192);
      launch_wcast(q.pw, 48, 192, 192);
      launch_gemm(abuf, 192, wbuf, 192, tmp1, 48, nullptr, q.pb, tok, 48, Mt, 48, 192, 0);
    }

    // final patch layernorm -> features f (tmp2)
    ln_kernel<<<(unsigned)Mt, 64, 0, stream>>>(tmp1, pat[bi].l2s, pat[bi].l2b, tmp2, nullptr, 48, 64);

    // capsule routing for this branch -> rows [OFFS..OFFS+n1) of concat buffer
    routing_kernel<<<64 * N1S[bi], 64, 0, stream>>>(tmp2, rW[bi], rb[bi], capc,
                                                    P, N1S[bi], 43, OFFS[bi]);
  }

  // final routing over concatenated capsules -> d_out [64, 10, 48]
  routing_kernel<<<64 * 10, 64, 0, stream>>>(capc, rW[4], rb[4], (float*)d_out, 43, 10, 10, 0);
}
